// KDABlock_57750130262726
// MI455X (gfx1250) — compile-verified, matrix operand order checked by reading, not run on hardware
//
#include <hip/hip_runtime.h>
#include <hip/hip_bf16.h>

typedef __attribute__((ext_vector_type(16))) __bf16 v16bf;
typedef __attribute__((ext_vector_type(8)))  __bf16 v8bf;
typedef __attribute__((ext_vector_type(4)))  __bf16 v4bf;
typedef __attribute__((ext_vector_type(8)))  float  v8f;

#define DEV_INLINE __device__ __forceinline__

static constexpr int Bc  = 8;
static constexpr int Tc  = 2048;
static constexpr int Dc  = 1024;
static constexpr int Hc  = 8;
static constexpr int DKc = 64;
static constexpr int DVc = 128;
static constexpr int Fc  = 4096;
static constexpr int CKc = 4;
static constexpr int BTc = Bc * Tc;
static constexpr float EPSc   = 1e-6f;
static constexpr float DECAYc = 0.9f;
static constexpr float THRc   = 0.5f;

DEV_INLINE float sigmoidf_(float x) { return 1.f / (1.f + __expf(-x)); }
DEV_INLINE float siluf_(float x)    { return x * sigmoidf_(x); }

// ---------------------------------------------------------------------------
// f32 -> bf16 conversion (weights / activations)
// ---------------------------------------------------------------------------
__global__ void cvt_f32_bf16_kernel(const float* __restrict__ in,
                                    __bf16* __restrict__ out, long n) {
  long i = (long)blockIdx.x * blockDim.x + threadIdx.x;
  long stride = (long)gridDim.x * blockDim.x;
  for (; i < n; i += stride) out[i] = (__bf16)in[i];
}

// ---------------------------------------------------------------------------
// 256-thread block reduction helper (8 waves of 32)
// ---------------------------------------------------------------------------
DEV_INLINE float block_reduce_sum_256(float v, float* sm) {
  #pragma unroll
  for (int o = 16; o > 0; o >>= 1) v += __shfl_xor(v, o);
  if ((threadIdx.x & 31) == 0) sm[threadIdx.x >> 5] = v;
  __syncthreads();
  float t = 0.f;
  #pragma unroll
  for (int i = 0; i < 8; ++i) t += sm[i];
  return t;
}

// ---------------------------------------------------------------------------
// RMSNorm stats: rstd per (b,t) row, and x -> bf16 copy
// ---------------------------------------------------------------------------
__global__ __launch_bounds__(256)
void rms_stats_kernel(const float* __restrict__ x, float* __restrict__ rstd,
                      __bf16* __restrict__ xbf) {
  __shared__ float sm[8];
  const int row = blockIdx.x;                 // b*T + t
  const float* xr = x + (size_t)row * Dc;
  float vals[4];
  float ss = 0.f;
  #pragma unroll
  for (int i = 0; i < 4; ++i) {
    int d = threadIdx.x + i * 256;
    float v = xr[d];
    vals[i] = v;
    ss += v * v;
  }
  ss = block_reduce_sum_256(ss, sm);
  float r = rsqrtf(ss / (float)Dc + EPSc);
  if (threadIdx.x == 0) rstd[row] = r;
  #pragma unroll
  for (int i = 0; i < 4; ++i) {
    int d = threadIdx.x + i * 256;
    xbf[(size_t)row * Dc + d] = (__bf16)vals[i];
  }
}

// ---------------------------------------------------------------------------
// Causal depthwise conv (CK=4) on rms-normed x, + SiLU -> h (bf16)
// ---------------------------------------------------------------------------
__global__ __launch_bounds__(256)
void conv_silu_kernel(const float* __restrict__ x, const float* __restrict__ rstd,
                      const float* __restrict__ wnorm, const float* __restrict__ convw,
                      const float* __restrict__ convb, __bf16* __restrict__ hbf) {
  const int row = blockIdx.x;
  const int b = row / Tc, t = row % Tc;
  for (int d = threadIdx.x; d < Dc; d += 256) {
    float acc = convb[d];
    #pragma unroll
    for (int j = 0; j < CKc; ++j) {
      int tt = t - (CKc - 1) + j;
      if (tt >= 0) {
        int r2 = b * Tc + tt;
        float hn = x[(size_t)r2 * Dc + d] * rstd[r2] * wnorm[d];
        acc += hn * convw[d * CKc + j];
      }
    }
    hbf[(size_t)row * Dc + d] = (__bf16)siluf_(acc);
  }
}

// ---------------------------------------------------------------------------
// bf16 WMMA GEMM: C[M,N] = A[M,K] @ B[K,N] (+bias) with epilogue.
// Tile 128x64, BK=32, 8 waves; A staged row-major, B staged transposed
// (both stride-40 rows so each fragment = 2x ds_load_b128).
// epi: 0=f32  1=bf16  2=silu->bf16  3=sigmoid->bf16  4=f32 accumulate-add
// ---------------------------------------------------------------------------
#define GBM 128
#define GBN 64
#define GBK 32
#define LDT 40   // padded LDS row stride in elements (80B: 16B-aligned, low-conflict)

union FragU { v8bf h[2]; v16bf f; };

DEV_INLINE void gemm_epilogue(v8f* acc, const float* bias, void* Cout,
                              int M, int N, int epi, int bm, int bn,
                              int wave, int half, int l15, bool guard) {
  #pragma unroll
  for (int nt = 0; nt < 4; ++nt) {
    int col = bn + nt * 16 + l15;
    if (guard && col >= N) continue;
    float bv = bias ? bias[col] : 0.f;
    #pragma unroll
    for (int r = 0; r < 8; ++r) {
      int row = bm + wave * 16 + r + 8 * half;
      if (guard && row >= M) continue;
      float v = acc[nt][r] + bv;
      size_t idx = (size_t)row * N + col;
      if (epi == 0)      ((float*)Cout)[idx]  = v;
      else if (epi == 1) ((__bf16*)Cout)[idx] = (__bf16)v;
      else if (epi == 2) ((__bf16*)Cout)[idx] = (__bf16)siluf_(v);
      else if (epi == 3) ((__bf16*)Cout)[idx] = (__bf16)sigmoidf_(v);
      else               ((float*)Cout)[idx] += v;
    }
  }
}

// Fast path: M % 128 == 0, N % 64 == 0. Double-buffered LDS, register-staged
// global fetch overlapped with WMMA compute; one barrier per k-step. All four
// B fragments preloaded so the WMMAs issue back-to-back.
__global__ __launch_bounds__(256)
void wmma_gemm_fast_kernel(const __bf16* __restrict__ A, const __bf16* __restrict__ Bm,
                           const float* __restrict__ bias, void* __restrict__ Cout,
                           int M, int N, int K, int epi) {
  __shared__ __bf16 As[2][GBM * LDT];
  __shared__ __bf16 Bt[2][GBN * LDT];
  const int tid  = threadIdx.x;
  const int wave = tid >> 5;
  const int lane = tid & 31;
  const int half = lane >> 4;
  const int l15  = lane & 15;
  const int bm = blockIdx.y * GBM;
  const int bn = blockIdx.x * GBN;

  v8f acc[4];
  #pragma unroll
  for (int nt = 0; nt < 4; ++nt) acc[nt] = (v8f){0.f,0.f,0.f,0.f,0.f,0.f,0.f,0.f};

  v4bf aReg[4], bReg[2];

  auto fetch = [&](int k0) {
    #pragma unroll
    for (int j = 0; j < 4; ++j) {            // A tile: 128x32 = 1024 chunks of 4
      int i = tid + j * 256;
      int r = i >> 3, c4 = (i & 7) * 4;
      aReg[j] = *reinterpret_cast<const v4bf*>(A + (size_t)(bm + r) * K + k0 + c4);
    }
    #pragma unroll
    for (int j = 0; j < 2; ++j) {            // B tile: 32x64 = 512 chunks of 4
      int i = tid + j * 256;
      int r = i >> 4, c4 = (i & 15) * 4;
      bReg[j] = *reinterpret_cast<const v4bf*>(Bm + (size_t)(k0 + r) * N + bn + c4);
    }
  };
  auto stage = [&](int buf) {
    #pragma unroll
    for (int j = 0; j < 4; ++j) {
      int i = tid + j * 256;
      int r = i >> 3, c4 = (i & 7) * 4;
      *reinterpret_cast<v4bf*>(&As[buf][r * LDT + c4]) = aReg[j];
    }
    #pragma unroll
    for (int j = 0; j < 2; ++j) {
      int i = tid + j * 256;
      int r = i >> 4, c4 = (i & 15) * 4;
      #pragma unroll
      for (int e = 0; e < 4; ++e) Bt[buf][(c4 + e) * LDT + r] = bReg[j][e];
    }
  };

  fetch(0);
  stage(0);
  int buf = 0;
  for (int k0 = 0; k0 < K; k0 += GBK) {
    const bool has_next = (k0 + GBK) < K;
    if (has_next) fetch(k0 + GBK);       // global loads in flight during compute
    __syncthreads();                     // LDS buf fully staged

    // Load A fragment + all four B fragments first (10x ds_load_b128),
    // then run the four WMMAs back-to-back.
    const __bf16* ap = As[buf] + (wave * 16 + l15) * LDT;
    FragU ua;
    ua.h[0] = *reinterpret_cast<const v8bf*>(ap + half * 8);
    ua.h[1] = *reinterpret_cast<const v8bf*>(ap + 16 + half * 8);
    FragU ub[4];
    #pragma unroll
    for (int nt = 0; nt < 4; ++nt) {
      const __bf16* bp = Bt[buf] + (nt * 16 + l15) * LDT;
      ub[nt].h[0] = *reinterpret_cast<const v8bf*>(bp + half * 8);
      ub[nt].h[1] = *reinterpret_cast<const v8bf*>(bp + 16 + half * 8);
    }
    #pragma unroll
    for (int nt = 0; nt < 4; ++nt)
      acc[nt] = __builtin_amdgcn_wmma_f32_16x16x32_bf16(
          false, ua.f, false, ub[nt].f, (short)0, acc[nt], false, false);

    if (has_next) stage(buf ^ 1);        // write the idle buffer
    buf ^= 1;
  }
  gemm_epilogue(acc, bias, Cout, M, N, epi, bm, bn, wave, half, l15, false);
}

// Guarded fallback for ragged N (only the N=8 beta projection here).
__global__ __launch_bounds__(256)
void wmma_gemm_guard_kernel(const __bf16* __restrict__ A, const __bf16* __restrict__ Bm,
                            const float* __restrict__ bias, void* __restrict__ Cout,
                            int M, int N, int K, int epi) {
  __shared__ __bf16 As[GBM * LDT];
  __shared__ __bf16 Bt[GBN * LDT];
  const int tid  = threadIdx.x;
  const int wave = tid >> 5;
  const int lane = tid & 31;
  const int half = lane >> 4;
  const int l15  = lane & 15;
  const int bm = blockIdx.y * GBM;
  const int bn = blockIdx.x * GBN;

  v8f acc[4];
  #pragma unroll
  for (int nt = 0; nt < 4; ++nt) acc[nt] = (v8f){0.f,0.f,0.f,0.f,0.f,0.f,0.f,0.f};
  const v4bf zero4 = {(__bf16)0.f, (__bf16)0.f, (__bf16)0.f, (__bf16)0.f};

  for (int k0 = 0; k0 < K; k0 += GBK) {
    #pragma unroll
    for (int i = tid; i < GBM * GBK / 4; i += 256) {
      int r = i >> 3, c4 = (i & 7) * 4;
      int gr = bm + r;
      v4bf val = zero4;
      if (gr < M) val = *reinterpret_cast<const v4bf*>(A + (size_t)gr * K + k0 + c4);
      *reinterpret_cast<v4bf*>(&As[r * LDT + c4]) = val;
    }
    #pragma unroll
    for (int i = tid; i < GBK * GBN / 4; i += 256) {
      int r = i >> 4, c4 = (i & 15) * 4;
      int gc = bn + c4;
      v4bf val = zero4;
      if (gc < N) val = *reinterpret_cast<const v4bf*>(Bm + (size_t)(k0 + r) * N + gc);
      #pragma unroll
      for (int j = 0; j < 4; ++j) Bt[(c4 + j) * LDT + r] = val[j];
    }
    __syncthreads();
    const __bf16* ap = As + (wave * 16 + l15) * LDT;
    FragU ua;
    ua.h[0] = *reinterpret_cast<const v8bf*>(ap + half * 8);
    ua.h[1] = *reinterpret_cast<const v8bf*>(ap + 16 + half * 8);
    #pragma unroll
    for (int nt = 0; nt < 4; ++nt) {
      const __bf16* bp = Bt + (nt * 16 + l15) * LDT;
      FragU ub;
      ub.h[0] = *reinterpret_cast<const v8bf*>(bp + half * 8);
      ub.h[1] = *reinterpret_cast<const v8bf*>(bp + 16 + half * 8);
      acc[nt] = __builtin_amdgcn_wmma_f32_16x16x32_bf16(
          false, ua.f, false, ub.f, (short)0, acc[nt], false, false);
    }
    __syncthreads();
  }
  gemm_epilogue(acc, bias, Cout, M, N, epi, bm, bn, wave, half, l15, true);
}

// ---------------------------------------------------------------------------
// L2 normalize rows of length 64 (q and k heads), in place, bf16
// ---------------------------------------------------------------------------
__global__ __launch_bounds__(64)
void l2norm64_kernel(__bf16* __restrict__ q) {
  __shared__ float p[2];
  const size_t row = blockIdx.x;
  const int lane = threadIdx.x;
  float v = (float)q[row * 64 + lane];
  float ss = v * v;
  #pragma unroll
  for (int o = 16; o > 0; o >>= 1) ss += __shfl_xor(ss, o);
  if ((lane & 31) == 0) p[lane >> 5] = ss;
  __syncthreads();
  float nrm = sqrtf(p[0] + p[1]) + EPSc;
  q[row * 64 + lane] = (__bf16)(v / nrm);
}

// ---------------------------------------------------------------------------
// Recurrent LIF + delta-rule scan. One block per (b,h); 128 threads; thread
// tid owns state column S[:,tid] (64 f32 registers). Outputs head-RMSNormed
// y (bf16) and final S / mem into d_out.
// ---------------------------------------------------------------------------
__global__ __launch_bounds__(128)
void scan_kernel(const __bf16* __restrict__ q, const __bf16* __restrict__ k,
                 const __bf16* __restrict__ v, const __bf16* __restrict__ sd,
                 const __bf16* __restrict__ ab, const float* __restrict__ bb,
                 const float* __restrict__ alpha_spike, const float* __restrict__ beta_spike,
                 const float* __restrict__ w_hn,
                 __bf16* __restrict__ yout, float* __restrict__ s_out,
                 float* __restrict__ mem_out) {
  const int bh = blockIdx.x;
  const int b = bh / Hc, h = bh % Hc;
  const int tid = threadIdx.x;                  // owns column v = tid

  __shared__ float kL[DKc], qL[DKc], aL[DKc];
  __shared__ float redA[2], redB[2], redO[4];

  float s[DKc];
  #pragma unroll
  for (int d = 0; d < DKc; ++d) s[d] = 0.f;
  float memv = 0.f;

  for (int t = 0; t < Tc; ++t) {
    const size_t basek = ((size_t)(b * Tc + t) * Hc + h) * DKc;
    if (tid < DKc) {
      float sdt = (float)sd[basek + tid];
      memv = DECAYc * memv + sdt;
      float hard = (memv > THRc) ? 1.f : 0.f;   // forward spike == hard
      memv -= hard * THRc;
      float ain = (float)ab[basek + tid] + alpha_spike[h * DKc + tid] * hard;
      aL[tid] = sigmoidf_(ain);
      kL[tid] = (float)k[basek + tid];
      qL[tid] = (float)q[basek + tid];
      float ph = hard, ps = hard * beta_spike[h * DKc + tid];
      #pragma unroll
      for (int o = 16; o > 0; o >>= 1) { ph += __shfl_xor(ph, o); ps += __shfl_xor(ps, o); }
      if ((tid & 31) == 0) { redA[tid >> 5] = ph; redB[tid >> 5] = ps; }
    }
    __syncthreads();

    float active = (redA[0] + redA[1] > 0.f) ? 1.f : 0.f;
    float beta = active * sigmoidf_(bb[(size_t)(b * Tc + t) * Hc + h] + redB[0] + redB[1]);

    float u = 0.f;
    #pragma unroll
    for (int d = 0; d < DKc; ++d) {
      float am = active * aL[d] + (1.f - active);   // alpha, or 1 if inactive
      s[d] *= am;
      u += kL[d] * s[d];
    }
    float vt = (float)v[((size_t)(b * Tc + t) * Hc + h) * DVc + tid];
    float delta = beta * (vt - u);
    float out = 0.f;
    #pragma unroll
    for (int d = 0; d < DKc; ++d) {
      s[d] += kL[d] * delta;
      out += qL[d] * s[d];
    }

    // headwise RMSNorm over DV=128 (4 waves)
    float o2 = out * out;
    #pragma unroll
    for (int o = 16; o > 0; o >>= 1) o2 += __shfl_xor(o2, o);
    if ((tid & 31) == 0) redO[tid >> 5] = o2;
    __syncthreads();
    float rstd = rsqrtf((redO[0] + redO[1] + redO[2] + redO[3]) / (float)DVc + EPSc);
    yout[(size_t)(b * Tc + t) * (Hc * DVc) + h * DVc + tid] =
        (__bf16)(out * rstd * w_hn[h * DVc + tid]);
  }

  // final state outputs
  #pragma unroll
  for (int d = 0; d < DKc; ++d)
    s_out[((size_t)(b * Hc + h) * DKc + d) * DVc + tid] = s[d];
  if (tid < DKc) mem_out[(size_t)(b * Hc + h) * DKc + tid] = memv;
}

// ---------------------------------------------------------------------------
// y1 = x + att*gate -> d_out(y); h2 = rms(y1)*w_ffn -> bf16
// ---------------------------------------------------------------------------
__global__ __launch_bounds__(256)
void gate_res_norm_kernel(const float* __restrict__ x, const float* __restrict__ att,
                          const __bf16* __restrict__ gate, const float* __restrict__ wffn,
                          float* __restrict__ yout, __bf16* __restrict__ h2bf) {
  __shared__ float sm[8];
  const int row = blockIdx.x;
  float vals[4];
  float ss = 0.f;
  #pragma unroll
  for (int i = 0; i < 4; ++i) {
    int d = threadIdx.x + i * 256;
    size_t idx = (size_t)row * Dc + d;
    float y = x[idx] + att[idx] * (float)gate[idx];
    vals[i] = y;
    ss += y * y;
  }
  ss = block_reduce_sum_256(ss, sm);
  float rstd = rsqrtf(ss / (float)Dc + EPSc);
  #pragma unroll
  for (int i = 0; i < 4; ++i) {
    int d = threadIdx.x + i * 256;
    size_t idx = (size_t)row * Dc + d;
    yout[idx] = vals[i];
    h2bf[idx] = (__bf16)(vals[i] * rstd * wffn[d]);
  }
}

// ---------------------------------------------------------------------------
// a *= b elementwise (bf16), for silu(ff1) * ff3
// ---------------------------------------------------------------------------
__global__ void mul_bf16_kernel(__bf16* __restrict__ a, const __bf16* __restrict__ b, long n) {
  long i = (long)blockIdx.x * blockDim.x + threadIdx.x;
  long stride = (long)gridDim.x * blockDim.x;
  for (; i < n; i += stride) a[i] = (__bf16)((float)a[i] * (float)b[i]);
}

// ---------------------------------------------------------------------------
// host launcher
// ---------------------------------------------------------------------------
extern "C" void kernel_launch(void* const* d_in, const int* in_sizes, int n_in,
                              void* d_out, int out_size, void* d_ws, size_t ws_size,
                              hipStream_t stream) {
  (void)in_sizes; (void)n_in; (void)out_size; (void)ws_size;
  const float* x        = (const float*)d_in[0];
  const float* w_norm   = (const float*)d_in[1];
  const float* conv_w   = (const float*)d_in[2];
  const float* conv_b   = (const float*)d_in[3];
  const float* wq       = (const float*)d_in[4];
  const float* wk       = (const float*)d_in[5];
  const float* wv       = (const float*)d_in[6];
  const float* w_out    = (const float*)d_in[7];
  const float* w_spike  = (const float*)d_in[8];
  const float* w_au     = (const float*)d_in[9];
  const float* b_au     = (const float*)d_in[10];
  const float* w_ad     = (const float*)d_in[11];
  const float* b_ad     = (const float*)d_in[12];
  const float* a_spk    = (const float*)d_in[13];
  const float* w_beta   = (const float*)d_in[14];
  const float* b_beta   = (const float*)d_in[15];
  const float* b_spk    = (const float*)d_in[16];
  const float* w_hn     = (const float*)d_in[17];
  const float* w_u1     = (const float*)d_in[18];
  const float* b_u1     = (const float*)d_in[19];
  const float* w_u2     = (const float*)d_in[20];
  const float* b_u2     = (const float*)d_in[21];
  const float* w_ffn    = (const float*)d_in[22];
  const float* w_ff1    = (const float*)d_in[23];
  const float* w_ff3    = (const float*)d_in[24];
  const float* w_ff2    = (const float*)d_in[25];

  float* yo      = (float*)d_out;                       // [BT, D]
  float* s_out   = yo + (size_t)BTc * Dc;               // [B,H,DK,DV]
  float* mem_out = s_out + (size_t)Bc * Hc * DKc * DVc; // [B,H,DK]

  // ---- workspace carve-up ----
  char* ws = (char*)d_ws;
  size_t off = 0;
  auto carve = [&](size_t bytes) -> void* {
    void* p = ws + off;
    off = (off + bytes + 255) & ~(size_t)255;
    return p;
  };
  __bf16* xbf    = (__bf16*)carve((size_t)BTc * Dc * 2);
  __bf16* hbf    = (__bf16*)carve((size_t)BTc * Dc * 2);
  __bf16* qbf    = (__bf16*)carve((size_t)BTc * Hc * DKc * 2);
  __bf16* kbf    = (__bf16*)carve((size_t)BTc * Hc * DKc * 2);
  __bf16* vbf    = (__bf16*)carve((size_t)BTc * Hc * DVc * 2);
  __bf16* sdbf   = (__bf16*)carve((size_t)BTc * Hc * DKc * 2);
  __bf16* aubf   = (__bf16*)carve((size_t)BTc * 64 * 2);
  __bf16* abbf   = (__bf16*)carve((size_t)BTc * Hc * DKc * 2);
  float*  bbf    = (float*)carve((size_t)BTc * Hc * 4);
  float*  rstd   = (float*)carve((size_t)BTc * 4);
  __bf16* ybf    = (__bf16*)carve((size_t)BTc * Hc * DVc * 2);
  __bf16* g1bf   = (__bf16*)carve((size_t)BTc * 64 * 2);
  __bf16* gatebf = (__bf16*)carve((size_t)BTc * Dc * 2);
  float*  att    = (float*)carve((size_t)BTc * Dc * 4);
  __bf16* h2bf   = (__bf16*)carve((size_t)BTc * Dc * 2);
  __bf16* f1bf   = (__bf16*)carve((size_t)BTc * Fc * 2);
  __bf16* f3bf   = (__bf16*)carve((size_t)BTc * Fc * 2);
  __bf16* wq_b   = (__bf16*)carve((size_t)Dc * 512 * 2);
  __bf16* wk_b   = (__bf16*)carve((size_t)Dc * 512 * 2);
  __bf16* wv_b   = (__bf16*)carve((size_t)Dc * 1024 * 2);
  __bf16* wsp_b  = (__bf16*)carve((size_t)Dc * 512 * 2);
  __bf16* wau_b  = (__bf16*)carve((size_t)Dc * 64 * 2);
  __bf16* wad_b  = (__bf16*)carve((size_t)64 * 512 * 2);
  __bf16* wbt_b  = (__bf16*)carve((size_t)Dc * 8 * 2);
  __bf16* wu1_b  = (__bf16*)carve((size_t)Dc * 64 * 2);
  __bf16* wu2_b  = (__bf16*)carve((size_t)64 * Dc * 2);
  __bf16* wout_b = (__bf16*)carve((size_t)Dc * Dc * 2);
  __bf16* wf1_b  = (__bf16*)carve((size_t)Dc * Fc * 2);
  __bf16* wf3_b  = (__bf16*)carve((size_t)Dc * Fc * 2);
  __bf16* wf2_b  = (__bf16*)carve((size_t)Fc * Dc * 2);

  auto cvt = [&](const float* src, __bf16* dst, long n) {
    int blocks = (int)((n + 255) / 256);
    if (blocks > 32768) blocks = 32768;
    cvt_f32_bf16_kernel<<<blocks, 256, 0, stream>>>(src, dst, n);
  };
  cvt(wq, wq_b, (long)Dc * 512);
  cvt(wk, wk_b, (long)Dc * 512);
  cvt(wv, wv_b, (long)Dc * 1024);
  cvt(w_spike, wsp_b, (long)Dc * 512);
  cvt(w_au, wau_b, (long)Dc * 64);
  cvt(w_ad, wad_b, (long)64 * 512);
  cvt(w_beta, wbt_b, (long)Dc * 8);
  cvt(w_u1, wu1_b, (long)Dc * 64);
  cvt(w_u2, wu2_b, (long)64 * Dc);
  cvt(w_out, wout_b, (long)Dc * Dc);
  cvt(w_ff1, wf1_b, (long)Dc * Fc);
  cvt(w_ff3, wf3_b, (long)Dc * Fc);
  cvt(w_ff2, wf2_b, (long)Fc * Dc);

  auto gemm = [&](const __bf16* A, const __bf16* Bm, const float* bias, void* C,
                  int M, int N, int K, int epi) {
    dim3 g((N + GBN - 1) / GBN, (M + GBM - 1) / GBM);
    if ((M % GBM == 0) && (N % GBN == 0))
      wmma_gemm_fast_kernel<<<g, 256, 0, stream>>>(A, Bm, bias, C, M, N, K, epi);
    else
      wmma_gemm_guard_kernel<<<g, 256, 0, stream>>>(A, Bm, bias, C, M, N, K, epi);
  };

  // 1) RMSNorm stats + x->bf16 ; conv + SiLU
  rms_stats_kernel<<<BTc, 256, 0, stream>>>(x, rstd, xbf);
  conv_silu_kernel<<<BTc, 256, 0, stream>>>(x, rstd, w_norm, conv_w, conv_b, hbf);

  // 2) projections (WMMA bf16)
  gemm(hbf, wq_b,  nullptr, qbf,  BTc, 512,  Dc, 1);
  gemm(hbf, wk_b,  nullptr, kbf,  BTc, 512,  Dc, 1);
  gemm(hbf, wv_b,  nullptr, vbf,  BTc, 1024, Dc, 1);
  gemm(hbf, wsp_b, nullptr, sdbf, BTc, 512,  Dc, 1);
  gemm(hbf, wbt_b, b_beta,  bbf,  BTc, 8,    Dc, 0);
  gemm(hbf, wau_b, b_au,    aubf, BTc, 64,   Dc, 2);   // silu
  gemm(aubf, wad_b, b_ad,   abbf, BTc, 512,  64, 1);

  // 3) normalize q,k per head (L2)
  l2norm64_kernel<<<(size_t)BTc * Hc, 64, 0, stream>>>(qbf);
  l2norm64_kernel<<<(size_t)BTc * Hc, 64, 0, stream>>>(kbf);

  // 4) sequential LIF + delta-rule scan (writes y bf16, and s/mem into d_out)
  scan_kernel<<<Bc * Hc, 128, 0, stream>>>(qbf, kbf, vbf, sdbf, abbf, bbf,
                                           a_spk, b_spk, w_hn, ybf, s_out, mem_out);

  // 5) gate chain + out-projection
  gemm(xbf,  wu1_b, b_u1, g1bf,   BTc, 64,   Dc, 2);   // silu
  gemm(g1bf, wu2_b, b_u2, gatebf, BTc, 1024, 64, 3);   // sigmoid
  gemm(ybf,  wout_b, nullptr, att, BTc, 1024, Dc, 0);  // f32

  // 6) y1 = x + att*gate -> d_out ; h2 = rms(y1)*w_ffn
  gate_res_norm_kernel<<<BTc, 256, 0, stream>>>(x, att, gatebf, w_ffn, yo, h2bf);

  // 7) FFN: silu(h2@w1) * (h2@w3) @ w2, accumulated into d_out y
  gemm(h2bf, wf1_b, nullptr, f1bf, BTc, Fc, Dc, 2);    // silu
  gemm(h2bf, wf3_b, nullptr, f3bf, BTc, Fc, Dc, 1);
  mul_bf16_kernel<<<32768, 256, 0, stream>>>(f1bf, f3bf, (long)BTc * Fc);
  gemm(f1bf, wf2_b, nullptr, yo, BTc, Dc, Fc, 4);      // += into y
}
